// GRU_57458072486093
// MI455X (gfx1250) — matrix operand order repaired and checked by reading, not compiled, not run on hardware
//
#include <hip/hip_runtime.h>
#include <hip/hip_bf16.h>

// GRU scan for MI455X (gfx1250, wave32, WMMA).
//  - 4 persistent workgroups, each owns a 16-row batch tile, full T-loop.
//  - Mixed precision: f16 WMMA (v_wmma_f32_16x16x32_f16), f32 accumulate.
//  - Weights pre-packed into WMMA B-fragment order in workspace (2.5 MB).
//  - A = [h | x_t] (K=768) folds input projection into recurrent GEMM.
//  - Output projection y = h @ W_hy deferred out of the serial loop:
//    h history (f16, 1024 B/row) aliases d_out (f32 y rows are also 1024 B),
//    then a chip-wide WMMA kernel converts h -> y in place.

typedef __attribute__((ext_vector_type(16))) _Float16 v16h;
typedef __attribute__((ext_vector_type(8)))  float    v8f;

#define B_  64
#define T_  1024
#define I_  256
#define H_  512
#define O_  256

#define KA_     768                     // H + I (combined K for gate GEMMs)
#define KT_G    24                      // 768/32 K-tiles (gates)
#define NT_G    32                      // 512/16 N-tiles (gates)
#define KT_Y    16                      // 512/32 K-tiles (output)
#define NT_Y    16                      // 256/16 N-tiles (output)
#define TILE_HALVES 512                 // one 32x16 B tile: 32 lanes * 16 halves
#define GATE_STRIDE ((size_t)KT_G * NT_G * TILE_HALVES)   // halves per gate
#define THREADS 512                     // 16 waves in the scan kernel

// ---------------------------------------------------------------------------
// Weight packing: f32 row-major W[k][n] -> f16 WMMA B-fragment order.
// B order used here (32x16 K x N, same k order as A): lane n (0-15) holds col n, K=0..15 in
// halves 0..15; lane n+16 holds col n, K=16..31.
// ---------------------------------------------------------------------------
__global__ void gru_pack_57458072486093(const float* __restrict__ Wz,
                                        const float* __restrict__ Wr,
                                        const float* __restrict__ Wh,
                                        const float* __restrict__ Why,
                                        _Float16* __restrict__ pg,
                                        _Float16* __restrict__ py) {
  const int idx = blockIdx.x * blockDim.x + threadIdx.x;
  const int gateWork = 3 * KT_G * NT_G * 32;   // tiles * lanes
  const int yWork    = KT_Y * NT_Y * 32;
  if (idx < gateWork) {
    const int lane = idx & 31;
    const int tile = idx >> 5;                 // 0..2303
    const int g  = tile / (KT_G * NT_G);
    const int r  = tile % (KT_G * NT_G);
    const int kt = r / NT_G, nt = r % NT_G;
    const float* W = (g == 0) ? Wz : (g == 1) ? Wr : Wh;
    const int kOff = kt * 32 + ((lane & 16) ? 16 : 0);
    const int n    = nt * 16 + (lane & 15);
    _Float16* dst = pg + (size_t)tile * TILE_HALVES + lane * 16;
#pragma unroll
    for (int j = 0; j < 16; ++j)
      dst[j] = (_Float16)W[(size_t)(kOff + j) * H_ + n];
  } else if (idx < gateWork + yWork) {
    const int idy  = idx - gateWork;
    const int lane = idy & 31;
    const int tile = idy >> 5;
    const int kt = tile / NT_Y, nt = tile % NT_Y;
    const int kOff = kt * 32 + ((lane & 16) ? 16 : 0);
    const int n    = nt * 16 + (lane & 15);
    _Float16* dst = py + (size_t)tile * TILE_HALVES + lane * 16;
#pragma unroll
    for (int j = 0; j < 16; ++j)
      dst[j] = (_Float16)Why[(size_t)(kOff + j) * O_ + n];
  }
}

// ---------------------------------------------------------------------------
// Fragment helpers.
// A order used here (16x32 M x K): lanes 0-15 row m=lane, VGPR j (0-7) -> K=2j,2j+1;
// lanes 16-31 same rows, K offset +16 (contiguous k, matches packed B).
// ---------------------------------------------------------------------------
__device__ __forceinline__ v16h load_a_frag(const _Float16* buf, int stride,
                                            int kBase, int lane) {
  const int m  = lane & 15;
  const int kb = kBase + ((lane & 16) ? 16 : 0);
  union { v16h v; unsigned u[8]; } f;
  const _Float16* row = buf + m * stride;
#pragma unroll
  for (int j = 0; j < 8; ++j) {
    const int k0 = kb + 2 * j;
    f.u[j] = *(const unsigned*)(row + k0);     // paired halves, dword LDS read
  }
  return f.v;
}

__device__ __forceinline__ v16h load_b_frag(const _Float16* tile, int lane) {
  return *(const v16h*)(tile + lane * 16);     // contiguous 32B per lane
}

__device__ __forceinline__ float sigmoid_f(float x) {
  return 1.0f / (1.0f + __expf(-x));
}
__device__ __forceinline__ float tanh_f(float x) {
  const float e = __expf(-2.0f * x);
  return (1.0f - e) / (1.0f + e);
}

// ---------------------------------------------------------------------------
// Persistent per-batch-tile GRU scan (gates only; y deferred).
// hHist aliases d_out: row r=(b*T+t) occupies bytes [r*1024, r*1024+1024).
// ---------------------------------------------------------------------------
__global__ void __launch_bounds__(THREADS, 1)
gru_scan_57458072486093(const float* __restrict__ x,
                        const float* __restrict__ bz, const float* __restrict__ br,
                        const float* __restrict__ bh,
                        const _Float16* __restrict__ pg,
                        _Float16* hHist) {
  // [m][0..511] = h (f16), [m][512..767] = x_t (f16)
  __shared__ __align__(16) _Float16 Abuf [16 * KA_];   // 24576 B
  __shared__ __align__(16) _Float16 RHbuf[16 * H_ ];   // 16384 B
  __shared__ __align__(16) _Float16 Zbuf [16 * H_ ];   // 16384 B

  const int tid   = threadIdx.x;
  const int lane  = tid & 31;
  const int wave  = tid >> 5;          // 0..15
  const int bBase = blockIdx.x * 16;
  const int n     = lane & 15;
  const int mB    = (lane & 16) ? 8 : 0;

  // Per-wave job assignment (fixed across the whole scan).
  const int gate = wave >> 3;                 // waves 0-7: z, 8-15: r
  const int ntB  = (wave & 7) * 4;            // 4 gate N-tiles per wave
  const _Float16* wgZR = pg + (size_t)gate * GATE_STRIDE;
  const _Float16* wgH  = pg + (size_t)2    * GATE_STRIDE;

  // Hoist per-lane gate biases (col = nt*16 + n is row-invariant in a C tile).
  const float* biasZR = gate ? br : bz;
  float bZR[4], bH[2];
#pragma unroll
  for (int i = 0; i < 4; ++i) bZR[i] = biasZR[(ntB + i) * 16 + n];
#pragma unroll
  for (int i = 0; i < 2; ++i) bH[i] = bh[(wave * 2 + i) * 16 + n];

  // h0 = 0
  for (int i = tid; i < 16 * H_; i += THREADS) {
    const int m = i >> 9, c = i & (H_ - 1);
    Abuf[m * KA_ + c] = (_Float16)0.0f;
  }
  __syncthreads();

  for (int t = 0; t < T_; ++t) {
    // ---- stage x_t into LDS (f32 -> f16), prefetch x_{t+1} ----
    for (int i = tid; i < 16 * I_; i += THREADS) {
      const int m = i >> 8, c = i & (I_ - 1);
      const size_t src = ((size_t)(bBase + m) * T_ + t) * I_ + c;
      Abuf[m * KA_ + H_ + c] = (_Float16)x[src];
      if (t + 1 < T_) __builtin_prefetch(&x[src + I_], 0, 1);
    }
    __syncthreads();                                      // B1

    // ---- phase ZR: z_pre = [h|x] @ W_z ; r_pre = [h|x] @ W_r ----
    {
      v8f acc[4] = {};
      for (int kt = 0; kt < KT_G; ++kt) {
        const v16h a = load_a_frag(Abuf, KA_, kt * 32, lane);
#pragma unroll
        for (int i = 0; i < 4; ++i) {
          const v16h b = load_b_frag(
              wgZR + (size_t)(kt * NT_G + ntB + i) * TILE_HALVES, lane);
          acc[i] = __builtin_amdgcn_wmma_f32_16x16x32_f16(
              false, a, false, b, (short)0, acc[i], false, false);
        }
      }
#pragma unroll
      for (int i = 0; i < 4; ++i) {
        const int nt = ntB + i;
#pragma unroll
        for (int e = 0; e < 8; ++e) {
          const int m = mB + e, col = nt * 16 + n;
          const float g = sigmoid_f(acc[i][e] + bZR[i]);
          if (gate == 0) {
            Zbuf[m * H_ + col] = (_Float16)g;
          } else {
            const float hOld = (float)Abuf[m * KA_ + col];
            RHbuf[m * H_ + col] = (_Float16)(g * hOld);
          }
        }
      }
    }
    __syncthreads();                                      // B2

    // ---- phase H: h~ = tanh([r*h | x] @ W_h + b_h); h = (1-z)h + z h~ ----
    {
      v8f acc[2] = {};
      for (int kt = 0; kt < KT_G; ++kt) {
        const v16h a = (kt < 16) ? load_a_frag(RHbuf, H_, kt * 32, lane)
                                 : load_a_frag(Abuf, KA_, kt * 32, lane);
#pragma unroll
        for (int i = 0; i < 2; ++i) {
          const v16h b = load_b_frag(
              wgH + (size_t)(kt * NT_G + wave * 2 + i) * TILE_HALVES, lane);
          acc[i] = __builtin_amdgcn_wmma_f32_16x16x32_f16(
              false, a, false, b, (short)0, acc[i], false, false);
        }
      }
#pragma unroll
      for (int i = 0; i < 2; ++i) {
        const int nt = wave * 2 + i;
#pragma unroll
        for (int e = 0; e < 8; ++e) {
          const int m = mB + e, col = nt * 16 + n;
          const float ht   = tanh_f(acc[i][e] + bH[i]);
          const float z    = (float)Zbuf[m * H_ + col];
          const float hOld = (float)Abuf[m * KA_ + col];
          const float hNew = (1.0f - z) * hOld + z * ht;
          const _Float16 h16 = (_Float16)hNew;
          Abuf[m * KA_ + col] = h16;                      // next-step A
          hHist[((size_t)(bBase + m) * T_ + t) * H_ + col] = h16;  // history
        }
      }
    }
    __syncthreads();                                      // B3
  }
}

// ---------------------------------------------------------------------------
// Chip-wide deferred output projection, in place over d_out.
// Block owns 16 flat (b,t) rows: stage h (f16) to LDS, GEMM, overwrite with y.
// No __restrict__ here: out/hSrc alias by design.
// ---------------------------------------------------------------------------
__global__ void __launch_bounds__(256)
gru_yproj_57458072486093(const _Float16* __restrict__ py,
                         const float* __restrict__ by,
                         const _Float16* hSrc, float* out) {
  __shared__ __align__(16) _Float16 Hs[16 * H_];         // 16 KB

  const int tid  = threadIdx.x;
  const int lane = tid & 31;
  const int wave = tid >> 5;                             // 0..7
  const size_t rBase = (size_t)blockIdx.x * 16;          // flat (b*T+t) rows
  const int n  = lane & 15;
  const int mB = (lane & 16) ? 8 : 0;

  // Stage 16 rows x 512 halves (reads the bytes we will later overwrite).
  const unsigned* src32 = (const unsigned*)(hSrc + rBase * H_);
  for (int i = tid; i < 16 * H_ / 2; i += 256)
    ((unsigned*)Hs)[i] = src32[i];
  __syncthreads();   // all h reads complete before any y store below

  v8f acc[2] = {};
  for (int kt = 0; kt < KT_Y; ++kt) {
    const v16h a = load_a_frag(Hs, H_, kt * 32, lane);
#pragma unroll
    for (int i = 0; i < 2; ++i) {
      const v16h b = load_b_frag(
          py + (size_t)(kt * NT_Y + wave * 2 + i) * TILE_HALVES, lane);
      acc[i] = __builtin_amdgcn_wmma_f32_16x16x32_f16(
          false, a, false, b, (short)0, acc[i], false, false);
    }
  }

#pragma unroll
  for (int i = 0; i < 2; ++i) {
    const int nt = wave * 2 + i;
    const float bias = by[nt * 16 + n];
#pragma unroll
    for (int e = 0; e < 8; ++e)
      out[(rBase + mB + e) * O_ + nt * 16 + n] = acc[i][e] + bias;
  }
}

// ---------------------------------------------------------------------------
extern "C" void kernel_launch(void* const* d_in, const int* in_sizes, int n_in,
                              void* d_out, int out_size, void* d_ws, size_t ws_size,
                              hipStream_t stream) {
  const float* x   = (const float*)d_in[0];
  const float* Wz  = (const float*)d_in[1];
  const float* bz  = (const float*)d_in[2];
  const float* Wr  = (const float*)d_in[3];
  const float* br  = (const float*)d_in[4];
  const float* Wh  = (const float*)d_in[5];
  const float* bh  = (const float*)d_in[6];
  const float* Why = (const float*)d_in[7];
  const float* by  = (const float*)d_in[8];
  float* out = (float*)d_out;

  _Float16* pg = (_Float16*)d_ws;                       // 3 gates packed
  _Float16* py = pg + 3 * GATE_STRIDE;                  // W_hy packed
  // workspace use: (3*393216 + 131072) halves = 2.5 MiB

  // h history aliases d_out: row (b*T+t) = 512 f16 = 1024 B = one y row.
  _Float16* hHist = (_Float16*)d_out;

  const int packWork = 3 * KT_G * NT_G * 32 + KT_Y * NT_Y * 32;   // 81920
  gru_pack_57458072486093<<<(packWork + 255) / 256, 256, 0, stream>>>(
      Wz, Wr, Wh, Why, pg, py);

  gru_scan_57458072486093<<<dim3(B_ / 16), dim3(THREADS), 0, stream>>>(
      x, bz, br, bh, pg, hHist);

  gru_yproj_57458072486093<<<dim3(B_ * T_ / 16), dim3(256), 0, stream>>>(
      py, by, hHist, out);
}
